// MoE_68453188764066
// MI455X (gfx1250) — matrix operand order repaired and checked by reading, not compiled, not run on hardware
//
#include <hip/hip_runtime.h>
#include <hip/hip_bf16.h>
#include <stddef.h>

typedef __bf16 bf16;
typedef __attribute__((ext_vector_type(16))) __bf16 v16bf;
typedef __attribute__((ext_vector_type(8)))  __bf16 v8bf;
typedef __attribute__((ext_vector_type(4)))  __bf16 v4bf;
typedef __attribute__((ext_vector_type(8)))  float  v8f;

#define NTOK 8192     // B*T = 4*2048
#define DIM  1024
#define NEXP 8
#define HID  4096
#define HB   512      // hidden K-block kept in LDS
#define XROWB ((DIM + 8) * 2)   // Xs row stride in bytes

// ---------------------------------------------------------------------------
// Kernel 0: zero output + workspace counters (out is accumulated atomically)
// ---------------------------------------------------------------------------
__global__ __launch_bounds__(256) void MoE_zero_kernel(float* __restrict__ out,
                                                       float* __restrict__ wshdr,
                                                       int nhdr) {
    size_t i = (size_t)blockIdx.x * blockDim.x + threadIdx.x;
    size_t stride = (size_t)gridDim.x * blockDim.x;
    size_t nv4 = ((size_t)NTOK * DIM) / 4;
    float4 z = make_float4(0.f, 0.f, 0.f, 0.f);
    for (size_t k = i; k < nv4; k += stride) ((float4*)out)[k] = z;
    for (size_t k = i; k < (size_t)nhdr; k += stride) wshdr[k] = 0.f;
    if (i == 0) out[(size_t)NTOK * DIM] = 0.f;   // balance-loss slot
}

// ---------------------------------------------------------------------------
// Kernel 0b: one-shot fp32 -> bf16 conversion of x / w_fc / w_proj into ws
// ---------------------------------------------------------------------------
__device__ __forceinline__ void cvt4(const float* __restrict__ s,
                                     bf16* __restrict__ d, size_t k) {
    float4 v = ((const float4*)s)[k];
    v4bf o;
    o[0] = (bf16)v.x; o[1] = (bf16)v.y; o[2] = (bf16)v.z; o[3] = (bf16)v.w;
    ((v4bf*)d)[k] = o;
}

__global__ __launch_bounds__(256) void MoE_cvt_kernel(
    const float* __restrict__ x, const float* __restrict__ w_fc,
    const float* __restrict__ w_proj, bf16* __restrict__ xbf,
    bf16* __restrict__ w1bf, bf16* __restrict__ w2bf) {
    size_t i = (size_t)blockIdx.x * blockDim.x + threadIdx.x;
    size_t stride = (size_t)gridDim.x * blockDim.x;
    size_t nx = ((size_t)NTOK * DIM) / 4;
    size_t nw = ((size_t)NEXP * HID * DIM) / 4;
    for (size_t k = i; k < nx; k += stride) cvt4(x, xbf, k);
    for (size_t k = i; k < nw; k += stride) { cvt4(w_fc, w1bf, k); cvt4(w_proj, w2bf, k); }
}

// ---------------------------------------------------------------------------
// Kernel 1: gating — logits, softmax, top-2, expert token lists, prob partials
// ---------------------------------------------------------------------------
__global__ __launch_bounds__(256) void MoE_gate_kernel(
    const float* __restrict__ x, const float* __restrict__ gate_w,
    int* __restrict__ cnt, float* __restrict__ partial,
    int* __restrict__ list, float* __restrict__ wlist) {
    __shared__ float gws[NEXP][DIM];      // 32 KB
    __shared__ float prsh[256][NEXP];     // 8 KB
    int tid = threadIdx.x;
    for (int i = tid; i < NEXP * DIM; i += 256) gws[i / DIM][i % DIM] = gate_w[i];
    __syncthreads();

    int n = blockIdx.x * 256 + tid;
    const float* xr = x + (size_t)n * DIM;
    float acc[NEXP];
#pragma unroll
    for (int e = 0; e < NEXP; ++e) acc[e] = 0.f;
    for (int d = 0; d < DIM; d += 4) {
        float4 xv = *(const float4*)(xr + d);
#pragma unroll
        for (int e = 0; e < NEXP; ++e) {
            acc[e] = fmaf(xv.x, gws[e][d + 0],
                     fmaf(xv.y, gws[e][d + 1],
                     fmaf(xv.z, gws[e][d + 2],
                     fmaf(xv.w, gws[e][d + 3], acc[e]))));
        }
    }
    float m = acc[0];
#pragma unroll
    for (int e = 1; e < NEXP; ++e) m = fmaxf(m, acc[e]);
    float p[NEXP], s = 0.f;
#pragma unroll
    for (int e = 0; e < NEXP; ++e) { p[e] = __expf(acc[e] - m); s += p[e]; }
    float inv = 1.f / s;
#pragma unroll
    for (int e = 0; e < NEXP; ++e) { p[e] *= inv; prsh[tid][e] = p[e]; }
    int i1 = 0; float v1 = p[0];
#pragma unroll
    for (int e = 1; e < NEXP; ++e) if (p[e] > v1) { v1 = p[e]; i1 = e; }
    int i2 = -1; float v2 = -1.f;
#pragma unroll
    for (int e = 0; e < NEXP; ++e) if (e != i1 && p[e] > v2) { v2 = p[e]; i2 = e; }
    float rs = 1.f / (v1 + v2);
    int pos1 = atomicAdd(&cnt[i1], 1);
    list[i1 * NTOK + pos1] = n;  wlist[i1 * NTOK + pos1] = v1 * rs;
    int pos2 = atomicAdd(&cnt[i2], 1);
    list[i2 * NTOK + pos2] = n;  wlist[i2 * NTOK + pos2] = v2 * rs;
    __syncthreads();
    if (tid < NEXP) {
        float sp = 0.f;
        for (int i = 0; i < 256; ++i) sp += prsh[i][tid];
        partial[blockIdx.x * NEXP + tid] = sp;
    }
}

// ---------------------------------------------------------------------------
// Kernel 2: balance loss = sum_e mean(prob_e) * freq_e * E  (fixed order)
// ---------------------------------------------------------------------------
__global__ void MoE_loss_kernel(const int* __restrict__ cnt,
                                const float* __restrict__ partial,
                                float* __restrict__ out_loss) {
    if (threadIdx.x == 0 && blockIdx.x == 0) {
        float loss = 0.f;
        for (int e = 0; e < NEXP; ++e) {
            float sp = 0.f;
            for (int b = 0; b < NTOK / 256; ++b) sp += partial[b * NEXP + e];
            loss += (sp * (1.f / NTOK)) * ((float)cnt[e] * (1.f / NTOK));
        }
        *out_loss = loss * (float)NEXP;
    }
}

// ---------------------------------------------------------------------------
// A/B fragment from row-major LDS (this code's k order, same for A and B):
// elems 0..15 at ap[0..15] (contiguous k)
// ---------------------------------------------------------------------------
__device__ __forceinline__ v16bf lds_ab(const bf16* ap) {
    v8bf lo = *(const v8bf*)ap;
    v8bf hi = *(const v8bf*)(ap + 8);
    return __builtin_shufflevector(lo, hi, 0, 1, 2, 3, 4, 5, 6, 7,
                                           8, 9, 10, 11, 12, 13, 14, 15);
}

#define WMMA_BF16(a, b, c) __builtin_amdgcn_wmma_f32_16x16x32_bf16( \
        false, (a), false, (b), (short)0, (c), false, false)

// ---------------------------------------------------------------------------
// Kernel 3 (fast path): fused expert MLP with pre-converted bf16 x/weights.
// One block = (expert, 16-token tile); 256 threads = 8 waves.
// X rows staged to LDS via CDNA5 async copy (ASYNCcnt).
// ---------------------------------------------------------------------------
__global__ __launch_bounds__(256) void MoE_mlp_bf16_kernel(
    const bf16* __restrict__ xbf, const bf16* __restrict__ w1bf,
    const bf16* __restrict__ w2bf, const int* __restrict__ cnt,
    const int* __restrict__ list, const float* __restrict__ wlist,
    float* __restrict__ out) {
    __shared__ bf16 Xs[16][DIM + 8];   // 33.0 KB
    __shared__ bf16 Hs[16][HB + 16];   // 16.9 KB
    __shared__ int   tok[16];
    __shared__ float wgt[16];

    const int e  = blockIdx.y;
    const int mt = blockIdx.x;
    const int cntE = cnt[e];
    if (mt * 16 >= cntE) return;

    const int tid  = threadIdx.x;
    const int wave = tid >> 5;
    const int lane = tid & 31;
    const int half = lane >> 4;
    const int lm   = lane & 15;

    if (tid < 16) {
        int idx = mt * 16 + tid;
        if (idx < cntE) { tok[tid] = list[e * NTOK + idx]; wgt[tid] = wlist[e * NTOK + idx]; }
        else            { tok[tid] = -1;                   wgt[tid] = 0.f; }
    }
    __syncthreads();

    // ---- async-stage 16 bf16 token rows (2 KB each) into LDS ----
    {
        int r  = tid >> 4;             // row 0..15
        int cb = (tid & 15) * 128;     // 128 bytes per thread per row
        int t  = tok[r];
        unsigned ldsoff = (unsigned)(size_t)(&Xs[0][0]) + (unsigned)(r * XROWB) + (unsigned)cb;
        if (t >= 0) {
            unsigned long long ga =
                (unsigned long long)(size_t)((const char*)(xbf + (size_t)t * DIM) + cb);
#pragma unroll
            for (int c = 0; c < 128; c += 16) {
                unsigned lo = ldsoff + (unsigned)c;
                unsigned long long g = ga + (unsigned long long)c;
                asm volatile("global_load_async_to_lds_b128 %0, %1, off"
                             :: "v"(lo), "v"(g) : "memory");
            }
        } else {
            bf16 z = (bf16)0.0f;
#pragma unroll
            for (int c = 0; c < 64; ++c) Xs[r][cb / 2 + c] = z;
        }
        asm volatile("s_wait_asynccnt 0x0" ::: "memory");
    }
    __syncthreads();

    const bf16* W1 = w1bf + (size_t)e * HID * DIM;  // [HID][DIM] bf16
    const bf16* W2 = w2bf + (size_t)e * DIM * HID;  // [DIM][HID] bf16

    v8f yacc[8];
#pragma unroll
    for (int j = 0; j < 8; ++j)
#pragma unroll
        for (int i = 0; i < 8; ++i) yacc[j][i] = 0.f;

    for (int hb = 0; hb < HID; hb += HB) {
        // ---- phase 1: Hs = relu(X @ W1[hb:hb+HB]^T)^2 ; K-outer, A reused 4x
        v8f hacc[4];
#pragma unroll
        for (int j = 0; j < 4; ++j)
#pragma unroll
            for (int i = 0; i < 8; ++i) hacc[j][i] = 0.f;
        for (int kk = 0; kk < DIM; kk += 32) {
            v16bf a = lds_ab(&Xs[lm][kk + (half << 4)]);
#pragma unroll
            for (int j = 0; j < 4; ++j) {
                int tcol = (j * 8 + wave) * 16;
                const bf16* bp = W1 + (size_t)(hb + tcol + lm) * DIM + kk + (half << 4);
                hacc[j] = WMMA_BF16(a, *(const v16bf*)bp, hacc[j]);
            }
        }
#pragma unroll
        for (int j = 0; j < 4; ++j) {
            int tcol = (j * 8 + wave) * 16;
#pragma unroll
            for (int i = 0; i < 8; ++i) {
                float r = fmaxf(hacc[j][i], 0.f);
                Hs[i + (half << 3)][tcol + lm] = (bf16)(r * r);
            }
        }
        __syncthreads();
        // ---- phase 2: yacc += Hs @ W2[:, hb:hb+HB]^T ; K-outer, A reused 8x
        for (int kk = 0; kk < HB; kk += 32) {
            v16bf a = lds_ab(&Hs[lm][kk + (half << 4)]);
#pragma unroll
            for (int j = 0; j < 8; ++j) {
                int ocol = (wave * 8 + j) * 16;
                const bf16* bp = W2 + (size_t)(ocol + lm) * HID + hb + kk + (half << 4);
                yacc[j] = WMMA_BF16(a, *(const v16bf*)bp, yacc[j]);
            }
        }
        __syncthreads();                                   // Hs reused next hb
    }

#pragma unroll
    for (int j = 0; j < 8; ++j) {
        int ocol = (wave * 8 + j) * 16 + lm;
#pragma unroll
        for (int i = 0; i < 8; ++i) {
            int m = i + (half << 3);
            int t = tok[m];
            if (t >= 0) atomicAdd(out + (size_t)t * DIM + ocol, yacc[j][i] * wgt[m]);
        }
    }
}

// ---------------------------------------------------------------------------
// Kernel 3 (fallback, small ws): fp32 weights with on-the-fly bf16 conversion
// ---------------------------------------------------------------------------
__global__ __launch_bounds__(256) void MoE_mlp_kernel(
    const float* __restrict__ x, const float* __restrict__ w_fc,
    const float* __restrict__ w_proj, const int* __restrict__ cnt,
    const int* __restrict__ list, const float* __restrict__ wlist,
    float* __restrict__ out) {
    __shared__ bf16 Xs[16][DIM + 8];
    __shared__ bf16 Hs[16][HB + 16];
    __shared__ int   tok[16];
    __shared__ float wgt[16];

    const int e  = blockIdx.y;
    const int mt = blockIdx.x;
    const int cntE = cnt[e];
    if (mt * 16 >= cntE) return;

    const int tid  = threadIdx.x;
    const int wave = tid >> 5;
    const int lane = tid & 31;
    const int half = lane >> 4;
    const int lm   = lane & 15;

    if (tid < 16) {
        int idx = mt * 16 + tid;
        if (idx < cntE) { tok[tid] = list[e * NTOK + idx]; wgt[tid] = wlist[e * NTOK + idx]; }
        else            { tok[tid] = -1;                   wgt[tid] = 0.f; }
    }
    __syncthreads();
    {
        int r = tid >> 4, cb = (tid & 15) * 64;
        int t = tok[r];
        const float* src = x + (size_t)(t < 0 ? 0 : t) * DIM + cb;
#pragma unroll 4
        for (int c = 0; c < 64; c += 4) {
            float4 v = (t >= 0) ? *(const float4*)(src + c) : make_float4(0.f, 0.f, 0.f, 0.f);
            Xs[r][cb + c + 0] = (bf16)v.x;  Xs[r][cb + c + 1] = (bf16)v.y;
            Xs[r][cb + c + 2] = (bf16)v.z;  Xs[r][cb + c + 3] = (bf16)v.w;
        }
    }
    __syncthreads();

    const float* W1 = w_fc   + (size_t)e * HID * DIM;
    const float* W2 = w_proj + (size_t)e * DIM * HID;

    v8f yacc[8];
#pragma unroll
    for (int j = 0; j < 8; ++j)
#pragma unroll
        for (int i = 0; i < 8; ++i) yacc[j][i] = 0.f;

    for (int hb = 0; hb < HID; hb += HB) {
#pragma unroll
        for (int tci = 0; tci < HB / 16 / 8; ++tci) {
            int tcol = (tci * 8 + wave) * 16;
            v8f acc;
#pragma unroll
            for (int i = 0; i < 8; ++i) acc[i] = 0.f;
            const float* brow = W1 + (size_t)(hb + tcol + lm) * DIM;
            for (int kk = 0; kk < DIM; kk += 32) {
                v16bf a = lds_ab(&Xs[lm][kk + (half << 4)]);
                const float* bp = brow + kk + (half << 4);
                __builtin_prefetch(bp + 64, 0, 0);
                v16bf b;
#pragma unroll
                for (int q = 0; q < 16; q += 4) {
                    float4 wv = *(const float4*)(bp + q);
                    b[q + 0] = (bf16)wv.x; b[q + 1] = (bf16)wv.y;
                    b[q + 2] = (bf16)wv.z; b[q + 3] = (bf16)wv.w;
                }
                acc = WMMA_BF16(a, b, acc);
            }
#pragma unroll
            for (int i = 0; i < 8; ++i) {
                float r = fmaxf(acc[i], 0.f);
                Hs[i + (half << 3)][tcol + lm] = (bf16)(r * r);
            }
        }
        __syncthreads();
#pragma unroll
        for (int j = 0; j < 8; ++j) {
            int ocol = (wave * 8 + j) * 16;
            const float* brow = W2 + (size_t)(ocol + lm) * HID + hb;
            v8f acc = yacc[j];
            for (int kk = 0; kk < HB; kk += 32) {
                v16bf a = lds_ab(&Hs[lm][kk + (half << 4)]);
                const float* bp = brow + kk + (half << 4);
                __builtin_prefetch(bp + 64, 0, 0);
                v16bf b;
#pragma unroll
                for (int q = 0; q < 16; q += 4) {
                    float4 wv = *(const float4*)(bp + q);
                    b[q + 0] = (bf16)wv.x; b[q + 1] = (bf16)wv.y;
                    b[q + 2] = (bf16)wv.z; b[q + 3] = (bf16)wv.w;
                }
                acc = WMMA_BF16(a, b, acc);
            }
            yacc[j] = acc;
        }
        __syncthreads();
    }
#pragma unroll
    for (int j = 0; j < 8; ++j) {
        int ocol = (wave * 8 + j) * 16 + lm;
#pragma unroll
        for (int i = 0; i < 8; ++i) {
            int m = i + (half << 3);
            int t = tok[m];
            if (t >= 0) atomicAdd(out + (size_t)t * DIM + ocol, yacc[j][i] * wgt[m]);
        }
    }
}

// ---------------------------------------------------------------------------
extern "C" void kernel_launch(void* const* d_in, const int* in_sizes, int n_in,
                              void* d_out, int out_size, void* d_ws, size_t ws_size,
                              hipStream_t stream) {
    const float* x      = (const float*)d_in[0];
    const float* gate_w = (const float*)d_in[1];
    const float* w_fc   = (const float*)d_in[2];
    const float* w_proj = (const float*)d_in[3];
    float* out = (float*)d_out;

    char* ws = (char*)d_ws;
    int*   cnt     = (int*)ws;                                    // 8 ints
    float* partial = (float*)(ws + 32);                           // 32*8 floats
    int*   list    = (int*)(ws + 32 + 1024);                      // E*N ints
    float* wlist   = (float*)(ws + 32 + 1024 + (size_t)NEXP * NTOK * 4);

    size_t base2 = 32 + 1024 + 2 * (size_t)NEXP * NTOK * 4;
    base2 = (base2 + 255) & ~(size_t)255;
    size_t xbf_sz = (size_t)NTOK * DIM * 2;
    size_t wbf_sz = (size_t)NEXP * HID * DIM * 2;
    size_t need = base2 + xbf_sz + 2 * wbf_sz;                    // ~151.5 MB
    bf16* xbf  = (bf16*)(ws + base2);
    bf16* w1bf = (bf16*)(ws + base2 + xbf_sz);
    bf16* w2bf = (bf16*)(ws + base2 + xbf_sz + wbf_sz);

    MoE_zero_kernel<<<2048, 256, 0, stream>>>(out, (float*)ws, 8 + 32 * NEXP);
    MoE_gate_kernel<<<NTOK / 256, 256, 0, stream>>>(x, gate_w, cnt, partial, list, wlist);
    MoE_loss_kernel<<<1, 32, 0, stream>>>(cnt, partial, out + (size_t)NTOK * DIM);

    dim3 grid(NTOK / 16, NEXP);
    if (ws_size >= need) {
        MoE_cvt_kernel<<<8192, 256, 0, stream>>>(x, w_fc, w_proj, xbf, w1bf, w2bf);
        MoE_mlp_bf16_kernel<<<grid, 256, 0, stream>>>(xbf, w1bf, w2bf, cnt, list, wlist, out);
    } else {
        MoE_mlp_kernel<<<grid, 256, 0, stream>>>(x, w_fc, w_proj, cnt, list, wlist, out);
    }
}
